// EnhancedDecoderAttention_56332791054981
// MI455X (gfx1250) — matrix operand order repaired and checked by reading, not compiled, not run on hardware
//
#include <hip/hip_runtime.h>
#include <hip/hip_bf16.h>

// ---------------------------------------------------------------------------
// EnhancedDecoderAttention for MI455X (gfx1250, wave32, WMMA bf16).
// Pipeline: convert -> QKV GEMMs (fused bias+RoPE / V-transpose epilogues)
//           -> flash attention (all WMMA, no LDS) -> output GEMM.
// GEMM wave tile: 32x64 (two row-tiles share B fragments -> ~21 FLOP/B).
// ---------------------------------------------------------------------------

typedef __attribute__((ext_vector_type(16))) __bf16 v16bf;
typedef __attribute__((ext_vector_type(8)))  float  v8f;
typedef __attribute__((ext_vector_type(4)))  unsigned int uint4v;
typedef __attribute__((ext_vector_type(4)))  float  float4v;

union BF16Frag {
    v16bf v;
    uint4v q[2];
    unsigned int d[8];
    unsigned short u[16];
};

__device__ __forceinline__ unsigned short f2bf(float f) {
    unsigned int u = __float_as_uint(f);
    unsigned int r = u + 0x7FFFu + ((u >> 16) & 1u);   // round-to-nearest-even
    return (unsigned short)(r >> 16);
}

#define SEQ  2048
#define EMB  1024
#define NH   16
#define HD   64
#define MROW 8192   // B * S

// ---------------------------------------------------------------------------
// fp32 -> bf16 elementwise convert
// ---------------------------------------------------------------------------
__global__ void cvt_bf16_kernel(const float* __restrict__ in,
                                unsigned short* __restrict__ out, int n) {
    int i = blockIdx.x * blockDim.x + threadIdx.x;
    if (i < n) out[i] = f2bf(in[i]);
}

// ---------------------------------------------------------------------------
// W[E][E] fp32 -> Wt[N][K] bf16 (transpose + convert)
// ---------------------------------------------------------------------------
__global__ void cvt_wt_kernel(const float* __restrict__ w,
                              unsigned short* __restrict__ wt) {
    int i = blockIdx.x * blockDim.x + threadIdx.x;   // 1M threads
    int n = i >> 10;
    int k = i & 1023;
    wt[(size_t)n * EMB + k] = f2bf(w[(size_t)k * EMB + n]);
}

// ---------------------------------------------------------------------------
// WMMA GEMM: C[M=8192, N=1024] = A_bf16[M][1024] * Bt_bf16[N][1024]^T + bias
// Block = 256 thr (8 waves). Wave tile: 32 rows x 64 cols (2 row-tiles x
// 4 N-tiles, B fragments shared across the two row-tiles).
// mode 0: bias + RoPE, write bf16 [B,H,S,D]      (Q and K)
// mode 2: bias, write bf16 transposed [B,H,D,S]  (V)
// mode 3: bias, write fp32 [M][E]                (output projection)
// ---------------------------------------------------------------------------
__global__ __launch_bounds__(256) void gemm_wmma_kernel(
    const unsigned short* __restrict__ A,
    const unsigned short* __restrict__ Bt,
    const float* __restrict__ bias,
    void* __restrict__ out, int mode)
{
    const int K = EMB;
    int lane = threadIdx.x & 31;
    int w    = threadIdx.x >> 5;
    int n16  = lane & 15;
    int h16  = lane >> 4;
    int row0 = blockIdx.x * 256 + w * 32;     // 32-row wave tile
    int col0 = blockIdx.y * 64;

    v8f acc[2][4];
#pragma unroll
    for (int rt = 0; rt < 2; ++rt)
#pragma unroll
        for (int c = 0; c < 4; ++c) acc[rt][c] = (v8f){};

    const unsigned short* arow0 = A  + (size_t)(row0 + n16) * K;
    const unsigned short* arow1 = A  + (size_t)(row0 + 16 + n16) * K;
    const unsigned short* brow  = Bt + (size_t)(col0 + n16) * K;

    for (int kk0 = 0; kk0 < K; kk0 += 32) {
        // prefetch one cacheline ahead on the streamed operands
        __builtin_prefetch(arow0 + kk0 + 64, 0, 1);
        __builtin_prefetch(arow1 + kk0 + 64, 0, 1);
        __builtin_prefetch(brow  + kk0 + 64, 0, 1);

        BF16Frag a0, a1;
        a0.q[0] = *(const uint4v*)(arow0 + kk0 + h16 * 8);
        a0.q[1] = *(const uint4v*)(arow0 + kk0 + 16 + h16 * 8);
        a1.q[0] = *(const uint4v*)(arow1 + kk0 + h16 * 8);
        a1.q[1] = *(const uint4v*)(arow1 + kk0 + 16 + h16 * 8);
#pragma unroll
        for (int c = 0; c < 4; ++c) {
            BF16Frag b;
            const unsigned short* bp = brow + (size_t)c * 16 * K + kk0 + h16 * 8;
            b.q[0] = *(const uint4v*)(bp);
            b.q[1] = *(const uint4v*)(bp + 16);
            acc[0][c] = __builtin_amdgcn_wmma_f32_16x16x32_bf16(
                false, a0.v, false, b.v, (short)0, acc[0][c], false, false);
            acc[1][c] = __builtin_amdgcn_wmma_f32_16x16x32_bf16(
                false, a1.v, false, b.v, (short)0, acc[1][c], false, false);
        }
    }

    int b    = row0 >> 11;        // row0 / 2048 (256-row blocks never cross b)
    int srow = row0 & 2047;
    int h    = col0 >> 6;         // one head per 64-col wave tile

    if (mode == 0) {
        // bias + RoPE -> bf16 [B,H,S,D]
        unsigned short* outp = (unsigned short*)out;
#pragma unroll
        for (int rt = 0; rt < 2; ++rt) {
            int srt = srow + rt * 16;
#pragma unroll
            for (int cp = 0; cp < 2; ++cp) {
                int   dl    = cp * 16 + n16;                       // in [0,32)
                float invf  = __powf(10000.0f, -((float)dl) * (1.0f / 32.0f));
                float bia_r = bias[col0 + cp * 16 + n16];
                float bia_i = bias[col0 + (cp + 2) * 16 + n16];
#pragma unroll
                for (int r = 0; r < 8; ++r) {
                    int   s  = srt + h16 * 8 + r;
                    float xr = acc[rt][cp][r] + bia_r;
                    float xi = acc[rt][cp + 2][r] + bia_i;
                    float ang = (float)s * invf;
                    float sn, cs;
                    sincosf(ang, &sn, &cs);
                    float vr = xr * cs - xi * sn;
                    float vi = xi * cs + xr * sn;
                    size_t base = (((size_t)(b * NH + h) * SEQ) + s) * HD + dl;
                    outp[base]      = f2bf(vr);
                    outp[base + 32] = f2bf(vi);
                }
            }
        }
    } else if (mode == 2) {
        // bias -> bf16 transposed [B,H,D,S] (key-contiguous rows)
        unsigned short* outp = (unsigned short*)out;
#pragma unroll
        for (int rt = 0; rt < 2; ++rt) {
            int srt = srow + rt * 16;
#pragma unroll
            for (int c = 0; c < 4; ++c) {
                int   dl = c * 16 + n16;
                float bi = bias[col0 + c * 16 + n16];
                uint4v pk;
#pragma unroll
                for (int rr = 0; rr < 4; ++rr) {
                    unsigned int lo = f2bf(acc[rt][c][2 * rr] + bi);
                    unsigned int hi = f2bf(acc[rt][c][2 * rr + 1] + bi);
                    pk[rr] = lo | (hi << 16);
                }
                size_t base = ((size_t)((b * NH + h) * HD + dl)) * SEQ + srt + h16 * 8;
                *(uint4v*)(outp + base) = pk;
            }
        }
    } else {
        // bias -> fp32 [M][E]
        float* outp = (float*)out;
#pragma unroll
        for (int rt = 0; rt < 2; ++rt) {
#pragma unroll
            for (int c = 0; c < 4; ++c) {
                float bi = bias[col0 + c * 16 + n16];
#pragma unroll
                for (int r = 0; r < 8; ++r) {
                    size_t row = (size_t)row0 + rt * 16 + h16 * 8 + r;
                    outp[row * EMB + col0 + c * 16 + n16] = acc[rt][c][r] + bi;
                }
            }
        }
    }
}

// ---------------------------------------------------------------------------
// Flash attention, one wave (32 thr) per (b, h, 16-query tile).
//   S^T  = K(16x64) . Q^T(64x16)  -> softmax over keys lives inside lanes
//   out^T = V^T(16x32) . P^T(32x16) accumulated over 32-key blocks
// P^T score registers -> B operand with one shfl_xor(16) per reg (no LDS).
// ---------------------------------------------------------------------------
__global__ __launch_bounds__(32) void attn_wmma_kernel(
    const unsigned short* __restrict__ qbf,   // [B,H,S,D] bf16 (rope'd)
    const unsigned short* __restrict__ kbf,   // [B,H,S,D] bf16 (rope'd)
    const unsigned short* __restrict__ vtb,   // [B,H,D,S] bf16
    float* __restrict__ O)                    // [B,S,E] fp32
{
    int lane = threadIdx.x & 31;
    int n16  = lane & 15;
    int h16  = lane >> 4;

    int idx = blockIdx.x;
    int qt  = idx & 127;            // S/16 = 128
    int h   = (idx >> 7) & 15;
    int b   = idx >> 11;
    int q0  = qt * 16;

    const unsigned short* qp = qbf + (size_t)(b * NH + h) * SEQ * HD;
    const unsigned short* kp = kbf + (size_t)(b * NH + h) * SEQ * HD;
    const unsigned short* vp = vtb + (size_t)(b * NH + h) * HD * SEQ;

    // Q as B operand (lane = query column, elems = d in table order: 8h+s, 16+8h+s)
    BF16Frag bq0, bq1;
    {
        const unsigned short* qrow = qp + (size_t)(q0 + n16) * HD;
        bq0.q[0] = *(const uint4v*)(qrow + h16 * 8);
        bq0.q[1] = *(const uint4v*)(qrow + 16 + h16 * 8);
        bq1.q[0] = *(const uint4v*)(qrow + 32 + h16 * 8);
        bq1.q[1] = *(const uint4v*)(qrow + 48 + h16 * 8);
    }

    v8f accv[4];
    accv[0] = (v8f){}; accv[1] = (v8f){}; accv[2] = (v8f){}; accv[3] = (v8f){};
    float Mrun = -1e30f, Lrun = 0.0f;
    int qidx = q0 + n16;

    int nblocks = (q0 + 47) >> 5;
    for (int blk = 0; blk < nblocks; ++blk) {
        int  t0   = blk * 32;
        bool last = (blk == nblocks - 1);

        // ---- scores S^T = K . Q^T  (two 16-key tiles) ----
        v8f st0 = (v8f){}, st1 = (v8f){};
        {
            const unsigned short* krow = kp + (size_t)(t0 + n16) * HD;
            BF16Frag a;
            a.q[0] = *(const uint4v*)(krow + h16 * 8);
            a.q[1] = *(const uint4v*)(krow + 16 + h16 * 8);
            st0 = __builtin_amdgcn_wmma_f32_16x16x32_bf16(
                false, a.v, false, bq0.v, (short)0, st0, false, false);
            a.q[0] = *(const uint4v*)(krow + 32 + h16 * 8);
            a.q[1] = *(const uint4v*)(krow + 48 + h16 * 8);
            st0 = __builtin_amdgcn_wmma_f32_16x16x32_bf16(
                false, a.v, false, bq1.v, (short)0, st0, false, false);
        }
        {
            const unsigned short* krow = kp + (size_t)(t0 + 16 + n16) * HD;
            BF16Frag a;
            a.q[0] = *(const uint4v*)(krow + h16 * 8);
            a.q[1] = *(const uint4v*)(krow + 16 + h16 * 8);
            st1 = __builtin_amdgcn_wmma_f32_16x16x32_bf16(
                false, a.v, false, bq0.v, (short)0, st1, false, false);
            a.q[0] = *(const uint4v*)(krow + 32 + h16 * 8);
            a.q[1] = *(const uint4v*)(krow + 48 + h16 * 8);
            st1 = __builtin_amdgcn_wmma_f32_16x16x32_bf16(
                false, a.v, false, bq1.v, (short)0, st1, false, false);
        }

        // ---- scale + causal mask (only the diagonal block needs it) ----
#pragma unroll
        for (int r = 0; r < 8; ++r) {
            float x0 = st0[r] * 0.125f;       // 1/sqrt(64)
            float x1 = st1[r] * 0.125f;
            if (last) {
                int k0 = t0 + h16 * 8 + r;
                int k1 = t0 + 16 + h16 * 8 + r;
                if (k0 > qidx) x0 = -1e30f;
                if (k1 > qidx) x1 = -1e30f;
            }
            st0[r] = x0;
            st1[r] = x1;
        }

        // ---- online softmax (key axis lives inside the lane) ----
        float bm = -1e30f;
#pragma unroll
        for (int r = 0; r < 8; ++r) {
            bm = fmaxf(bm, st0[r]);
            bm = fmaxf(bm, st1[r]);
        }
        bm = fmaxf(bm, __shfl_xor(bm, 16, 32));
        float newM  = fmaxf(Mrun, bm);
        float alpha = __expf(Mrun - newM);
        Mrun = newM;

        float ps = 0.0f;
        unsigned short pb0[8], pb1[8];
#pragma unroll
        for (int r = 0; r < 8; ++r) {
            float p0 = __expf(st0[r] - newM);
            float p1 = __expf(st1[r] - newM);
            ps += p0 + p1;
            pb0[r] = f2bf(p0);
            pb1[r] = f2bf(p1);
        }
        ps += __shfl_xor(ps, 16, 32);
        Lrun = Lrun * alpha + ps;
#pragma unroll
        for (int c = 0; c < 4; ++c)
#pragma unroll
            for (int r = 0; r < 8; ++r) accv[c][r] *= alpha;

        // ---- P^T (C-layout) -> B operand: one shfl_xor(16) per reg ----
        BF16Frag bp;
#pragma unroll
        for (int r = 0; r < 8; ++r) {
            unsigned int o0 = (unsigned int)__shfl_xor((int)pb0[r], 16, 32);
            unsigned int o1 = (unsigned int)__shfl_xor((int)pb1[r], 16, 32);
            bp.u[r]     = (lane < 16) ? pb0[r] : (unsigned short)o1;  // key t0+h16*16+r
            bp.u[8 + r] = (lane < 16) ? (unsigned short)o0 : pb1[r];  // key t0+h16*16+8+r
        }

        // ---- out^T += V^T . P^T  (4 d-tiles of 16) ----
#pragma unroll
        for (int c = 0; c < 4; ++c) {
            const unsigned short* vrow = vp + (size_t)(c * 16 + n16) * SEQ + t0;
            BF16Frag a;
            a.q[0] = *(const uint4v*)(vrow + h16 * 16);
            a.q[1] = *(const uint4v*)(vrow + h16 * 16 + 8);
            accv[c] = __builtin_amdgcn_wmma_f32_16x16x32_bf16(
                false, a.v, false, bp.v, (short)0, accv[c], false, false);
        }
    }

    // ---- normalize and write O[b, s, h*64 + d] ----
    float rl = 1.0f / Lrun;
    int   sq = q0 + n16;
    float* orow = O + ((size_t)b * SEQ + sq) * EMB + h * HD;
#pragma unroll
    for (int c = 0; c < 4; ++c) {
        float4v lo, hi;
#pragma unroll
        for (int r = 0; r < 4; ++r) {
            lo[r] = accv[c][r] * rl;
            hi[r] = accv[c][4 + r] * rl;
        }
        int dbase = c * 16 + h16 * 8;
        *(float4v*)(orow + dbase)     = lo;
        *(float4v*)(orow + dbase + 4) = hi;
    }
}

// ---------------------------------------------------------------------------
// Launch
// ---------------------------------------------------------------------------
extern "C" void kernel_launch(void* const* d_in, const int* in_sizes, int n_in,
                              void* d_out, int out_size, void* d_ws, size_t ws_size,
                              hipStream_t stream) {
    (void)in_sizes; (void)n_in; (void)out_size; (void)ws_size;
    const float* x  = (const float*)d_in[0];
    // d_in[1] = causal mask (implied by kernel, unused)
    const float* Wq = (const float*)d_in[2];
    const float* bq = (const float*)d_in[3];
    const float* Wk = (const float*)d_in[4];
    const float* bk = (const float*)d_in[5];
    const float* Wv = (const float*)d_in[6];
    const float* bv = (const float*)d_in[7];
    const float* Wo = (const float*)d_in[8];
    const float* bo = (const float*)d_in[9];

    char* ws = (char*)d_ws;
    const size_t MB = 1024 * 1024;
    unsigned short* xbf = (unsigned short*)(ws + 0);        // 16 MB
    unsigned short* wtq = (unsigned short*)(ws + 16 * MB);  // 2 MB
    unsigned short* wtk = (unsigned short*)(ws + 18 * MB);  // 2 MB
    unsigned short* wtv = (unsigned short*)(ws + 20 * MB);  // 2 MB
    unsigned short* wto = (unsigned short*)(ws + 22 * MB);  // 2 MB
    unsigned short* qbf = (unsigned short*)(ws + 24 * MB);  // 16 MB
    unsigned short* kbf = (unsigned short*)(ws + 40 * MB);  // 16 MB
    unsigned short* vtb = (unsigned short*)(ws + 56 * MB);  // 16 MB
    float*          Obf = (float*)(ws + 72 * MB);           // 32 MB
    unsigned short* obf = qbf;  // reuse Q buffer after attention; total 104 MB

    const int NTOK = MROW * EMB;               // 8.4M elements
    dim3 cgrid((NTOK + 255) / 256);
    cvt_bf16_kernel<<<cgrid, 256, 0, stream>>>(x, xbf, NTOK);

    dim3 tgrid((EMB * EMB) / 256);
    cvt_wt_kernel<<<tgrid, 256, 0, stream>>>(Wq, wtq);
    cvt_wt_kernel<<<tgrid, 256, 0, stream>>>(Wk, wtk);
    cvt_wt_kernel<<<tgrid, 256, 0, stream>>>(Wv, wtv);
    cvt_wt_kernel<<<tgrid, 256, 0, stream>>>(Wo, wto);

    dim3 ggrid(MROW / 256, EMB / 64);          // (32, 16)
    gemm_wmma_kernel<<<ggrid, 256, 0, stream>>>(xbf, wtq, bq, qbf, 0);
    gemm_wmma_kernel<<<ggrid, 256, 0, stream>>>(xbf, wtk, bk, kbf, 0);
    gemm_wmma_kernel<<<ggrid, 256, 0, stream>>>(xbf, wtv, bv, vtb, 2);

    attn_wmma_kernel<<<4 * NH * (SEQ / 16), 32, 0, stream>>>(qbf, kbf, vtb, Obf);

    cvt_bf16_kernel<<<cgrid, 256, 0, stream>>>(Obf, obf, NTOK);
    gemm_wmma_kernel<<<ggrid, 256, 0, stream>>>(obf, wto, bo, d_out, 3);
}
